// Attention_29515015258389
// MI455X (gfx1250) — matrix operand order repaired and checked by reading, not compiled, not run on hardware
//
#include <hip/hip_runtime.h>
#include <hip/hip_bf16.h>
#include <math.h>
#include <stdint.h>

typedef __bf16 bf16_t;
typedef __attribute__((ext_vector_type(16))) __bf16 v16bf;
typedef __attribute__((ext_vector_type(8)))  __bf16 v8bf;
typedef __attribute__((ext_vector_type(4)))  __bf16 v4bf;
typedef __attribute__((ext_vector_type(8)))  float  v8f;
typedef __attribute__((ext_vector_type(4)))  unsigned v4u;
typedef __attribute__((ext_vector_type(8)))  unsigned v8u;

#define DIMC   1024
#define NHEADS 16
#define HDIM   64
#define SEQ    2048
#define BATCH  4
#define NTOK   (BATCH * SEQ)   // 8192

// ---------------------------------------------------------------------------
// CDNA5 async copy global->LDS (ASYNCcnt) + wait.
// ---------------------------------------------------------------------------
__device__ inline void async_ld_b128(void* lds_dst, const void* gsrc) {
  unsigned l = (unsigned)(uintptr_t)lds_dst;
  asm volatile("global_load_async_to_lds_b128 %0, %1, off"
               :: "v"(l), "v"(gsrc) : "memory");
}
__device__ inline void wait_async0() {
  asm volatile("s_wait_asynccnt 0x0" ::: "memory");
}

// ---------------------------------------------------------------------------
// TDM: 2D bf16 tile global -> LDS (TENSORcnt). Descriptor per ISA 8.3/8.4.
// rows x cols elements, global row stride in elements. Issue from ONE wave.
// ---------------------------------------------------------------------------
__device__ inline void tdm_load_tile_2d(void* lds_dst, const bf16_t* gsrc,
                                        unsigned rows, unsigned cols,
                                        unsigned row_stride_elems) {
  unsigned long long ga = (unsigned long long)(uintptr_t)gsrc;
  v4u g0;
  g0[0] = 1u;                                   // count=1, user descriptor
  g0[1] = (unsigned)(uintptr_t)lds_dst;         // lds_addr
  g0[2] = (unsigned)(ga & 0xFFFFFFFFu);         // global_addr[31:0]
  g0[3] = (unsigned)((ga >> 32) & 0x01FFFFFFu)  // global_addr[56:32]
        | 0x80000000u;                          // type=2 ("image")
  v8u g1;
  g1[0] = 0x00010000u;                          // data_size=1 (2 bytes)
  g1[1] = cols << 16;                           // tensor_dim0 low16 (bits 63:48)
  g1[2] = rows << 16;                           // dim0 hi=0 | tensor_dim1 low16
  g1[3] = cols << 16;                           // dim1 hi=0 | tile_dim0
  g1[4] = rows;                                 // tile_dim1 | tile_dim2=0
  g1[5] = row_stride_elems;                     // tensor_dim0_stride low32
  g1[6] = 0u;                                   // stride0 hi | stride1 low
  g1[7] = 0u;                                   // stride1 hi
  asm volatile("tensor_load_to_lds %0, %1, null, null"
               :: "s"(g0), "s"(g1) : "memory");
}

// ---------------------------------------------------------------------------
// DPP row_xmask butterfly reductions within 16-lane halves (no LDS traffic)
// ---------------------------------------------------------------------------
template <int N>
__device__ inline float dpp_xor_f(float x) {
  int y = __builtin_amdgcn_update_dpp(0, __float_as_int(x), 0x160 + N, 0xF, 0xF, true);
  return __int_as_float(y);
}
__device__ inline float rowmax16(float x) {
  x = fmaxf(x, dpp_xor_f<1>(x));
  x = fmaxf(x, dpp_xor_f<2>(x));
  x = fmaxf(x, dpp_xor_f<4>(x));
  x = fmaxf(x, dpp_xor_f<8>(x));
  return x;
}
__device__ inline float rowsum16(float x) {
  x += dpp_xor_f<1>(x);
  x += dpp_xor_f<2>(x);
  x += dpp_xor_f<4>(x);
  x += dpp_xor_f<8>(x);
  return x;
}

// ---------------------------------------------------------------------------
// Fragment loaders (wave32 WMMA layouts per CDNA5 ISA 7.12.2)
// ---------------------------------------------------------------------------
__device__ inline v16bf lds_afrag(const bf16_t* p) {
  v8bf lo = *reinterpret_cast<const v8bf*>(p);
  v8bf hi = *reinterpret_cast<const v8bf*>(p + 16);
  v16bf f;
#pragma unroll
  for (int j = 0; j < 8; ++j) { f[j] = lo[j]; f[8 + j] = hi[j]; }
  return f;
}
__device__ inline v16bf lds_bfrag(const bf16_t* p) {
  v8bf lo = *reinterpret_cast<const v8bf*>(p);
  v8bf hi = *reinterpret_cast<const v8bf*>(p + 16);
  v16bf f;
#pragma unroll
  for (int j = 0; j < 8; ++j) { f[j] = lo[j]; f[8 + j] = hi[j]; }
  return f;
}

// ---------------------------------------------------------------------------
// f32 -> bf16 pack
// ---------------------------------------------------------------------------
__global__ __launch_bounds__(256) void pack_bf16_kernel(
    const float* __restrict__ in, bf16_t* __restrict__ out, int n4)
{
  int i = blockIdx.x * 256 + threadIdx.x;
  if (i >= n4) return;
  float4 f = reinterpret_cast<const float4*>(in)[i];
  v4bf o;
  o[0] = (bf16_t)f.x; o[1] = (bf16_t)f.y; o[2] = (bf16_t)f.z; o[3] = (bf16_t)f.w;
  reinterpret_cast<v4bf*>(out)[i] = o;
}

// ---------------------------------------------------------------------------
// GEMM + bias: C[M,N] = A @ B + bias; A/B bf16; 4 waves; WG tile 64x128.
// A staged via async-to-LDS, B transposed into LDS.
// ---------------------------------------------------------------------------
template <typename OutT>
__global__ __launch_bounds__(128) void gemm_bias_kernel(
    const bf16_t* __restrict__ A, const bf16_t* __restrict__ B,
    const float* __restrict__ bias, OutT* __restrict__ C,
    int M, int N, int K)
{
  __shared__ __align__(32) bf16_t sA [64][32];
  __shared__ __align__(32) bf16_t sBt[128][32];

  const int tid  = threadIdx.x;
  const int lane = tid & 31;
  const int wave = tid >> 5;
  const int half = lane >> 4;
  const int l16  = lane & 15;
  const int wm   = wave >> 1;
  const int wn   = wave & 1;
  const size_t bm = (size_t)blockIdx.y * 64;
  const size_t bn = (size_t)blockIdx.x * 128;

  v8f acc[2][4] = {};

  const int arow = tid >> 1;
  const int acol = (tid & 1) * 16;
  const int brow = tid >> 2;
  const int bnb  = (tid & 3) * 32;

  for (int k0 = 0; k0 < K; k0 += 32) {
    const bf16_t* ap = A + (bm + arow) * (size_t)K + k0 + acol;
    async_ld_b128(&sA[arow][acol],     ap);
    async_ld_b128(&sA[arow][acol + 8], ap + 8);
    __builtin_prefetch(ap + 32, 0, 0);

    const bf16_t* bp = B + (size_t)(k0 + brow) * N + bn + bnb;
#pragma unroll
    for (int j = 0; j < 4; ++j) {
      v8bf g = *reinterpret_cast<const v8bf*>(bp + 8 * j);
#pragma unroll
      for (int e = 0; e < 8; ++e) sBt[bnb + 8 * j + e][brow] = g[e];
    }
    __builtin_prefetch(bp + (size_t)32 * N, 0, 0);

    wait_async0();
    __syncthreads();

    v16bf af[2], bfr[4];
#pragma unroll
    for (int mi = 0; mi < 2; ++mi)
      af[mi] = lds_afrag(&sA[wm * 32 + mi * 16 + l16][half * 8]);
#pragma unroll
    for (int ni = 0; ni < 4; ++ni)
      bfr[ni] = lds_bfrag(&sBt[wn * 64 + ni * 16 + l16][half * 8]);

#pragma unroll
    for (int mi = 0; mi < 2; ++mi)
#pragma unroll
      for (int ni = 0; ni < 4; ++ni)
        acc[mi][ni] = __builtin_amdgcn_wmma_f32_16x16x32_bf16(
            false, af[mi], false, bfr[ni], (short)0, acc[mi][ni], false, false);
    __syncthreads();
  }

#pragma unroll
  for (int mi = 0; mi < 2; ++mi)
#pragma unroll
    for (int ni = 0; ni < 4; ++ni) {
      size_t col = bn + wn * 64 + ni * 16 + l16;
      float bv = bias[col];
#pragma unroll
      for (int e = 0; e < 8; ++e) {
        size_t row = bm + wm * 32 + mi * 16 + e + 8 * half;
        C[row * (size_t)N + col] = (OutT)(acc[mi][ni][e] + bv);
      }
    }
}

// ---------------------------------------------------------------------------
// RoPE in-place on q,k of bf16 qkv (NTOK x 3*DIMC)
// ---------------------------------------------------------------------------
__global__ __launch_bounds__(256) void rope_kernel(bf16_t* __restrict__ qkv)
{
  int idx = blockIdx.x * 256 + threadIdx.x;
  int i   = idx & 31;
  int h   = (idx >> 5) & 15;
  int qk  = (idx >> 9) & 1;
  int tok = idx >> 10;
  int t   = tok & (SEQ - 1);
  float inv_freq = __powf(10000.0f, -(float)(2 * i) * (1.0f / 64.0f));
  float ang = (float)t * inv_freq;
  float s, c;
  __sincosf(ang, &s, &c);
  bf16_t* p = qkv + (size_t)tok * (3 * DIMC) + qk * DIMC + h * HDIM;
  float x1 = (float)p[i], x2 = (float)p[i + 32];
  p[i]      = (bf16_t)(x1 * c - x2 * s);
  p[i + 32] = (bf16_t)(x2 * c + x1 * s);
}

// ---------------------------------------------------------------------------
// Flash attention: grid (SEQ/64, NHEADS, BATCH), 128 thr = 4 waves.
// K/V tiles of 64 rows. K tile fetched by the Tensor Data Mover (wave 0),
// V transposed while staging. 16 WMMAs per iteration.
// ---------------------------------------------------------------------------
__global__ __launch_bounds__(128) void attn_kernel(
    const bf16_t* __restrict__ qkv, bf16_t* __restrict__ out)
{
  __shared__ __align__(32) bf16_t sK [64][64];     // K tile [krow][dim]  (TDM)
  __shared__ __align__(32) bf16_t sVt[64][64];     // V tile transposed [dim][krow]
  __shared__ __align__(32) bf16_t sP [4][16][64];  // per-wave P staging

  const int tid  = threadIdx.x;
  const int lane = tid & 31;
  const int wave = tid >> 5;
  const int half = lane >> 4;
  const int l16  = lane & 15;

  const int b  = blockIdx.z;
  const int h  = blockIdx.y;
  const int q0 = blockIdx.x * 64;
  const float scale = 0.125f;   // 1/sqrt(64)

  // q fragments (A layout, 2 K-chunks of 32)
  const size_t tok_q = (size_t)b * SEQ + q0 + wave * 16 + l16;
  const bf16_t* qp = qkv + tok_q * (3 * DIMC) + h * HDIM;
  v16bf qf[2];
#pragma unroll
  for (int c = 0; c < 2; ++c) {
    v8bf lo = *reinterpret_cast<const v8bf*>(qp + c * 32 + half * 8);
    v8bf hi = *reinterpret_cast<const v8bf*>(qp + c * 32 + 16 + half * 8);
#pragma unroll
    for (int j = 0; j < 8; ++j) { qf[c][j] = lo[j]; qf[c][8 + j] = hi[j]; }
  }

  float rmax[8], rsum[8];                 // rsum: lane-partial (deferred reduce)
#pragma unroll
  for (int e = 0; e < 8; ++e) { rmax[e] = -1e30f; rsum[e] = 0.0f; }
  v8f oacc[4] = {};

  const int vrow = tid >> 1;              // V staging: 0..63
  const int vch  = (tid & 1) * 32;

  for (int kt = 0; kt < SEQ / 64; ++kt) {
    // --- K tile via Tensor Data Mover (one wave issues; EXEC-independent) ---
    if (tid < 32) {
      const bf16_t* kbase =
          qkv + ((size_t)b * SEQ + kt * 64) * (3 * DIMC) + DIMC + h * HDIM;
      tdm_load_tile_2d(&sK[0][0], kbase, 64u, 64u, 3u * DIMC);
    }

    // --- V tile: transpose while staging ---
    {
      const size_t tok_v = (size_t)b * SEQ + kt * 64 + vrow;
      const bf16_t* vp = qkv + tok_v * (3 * DIMC) + 2 * DIMC + h * HDIM + vch;
#pragma unroll
      for (int jj = 0; jj < 4; ++jj) {
        v8bf g = *reinterpret_cast<const v8bf*>(vp + 8 * jj);
#pragma unroll
        for (int j = 0; j < 8; ++j) sVt[vch + 8 * jj + j][vrow] = g[j];
      }
      __builtin_prefetch(vp + (size_t)64 * 3 * DIMC, 0, 0);
    }

    if (tid < 32) __builtin_amdgcn_s_wait_tensorcnt(0);
    __syncthreads();

    // --- S = q @ k^T : 16x64 score tile (4 N-subtiles x 2 K-chunks) ---
    v8f s[4] = {};
#pragma unroll
    for (int c = 0; c < 2; ++c)
#pragma unroll
      for (int g = 0; g < 4; ++g) {
        v16bf kf = lds_bfrag(&sK[g * 16 + l16][c * 32 + half * 8]);
        s[g] = __builtin_amdgcn_wmma_f32_16x16x32_bf16(
            false, qf[c], false, kf, (short)0, s[g], false, false);
      }

    // --- online softmax: DPP max reduce; sum kept lane-partial ---
    float p[4][8], corr[8];
#pragma unroll
    for (int e = 0; e < 8; ++e) {
      float sc0 = s[0][e] * scale, sc1 = s[1][e] * scale;
      float sc2 = s[2][e] * scale, sc3 = s[3][e] * scale;
      float mx = fmaxf(fmaxf(sc0, sc1), fmaxf(sc2, sc3));
      mx = rowmax16(mx);
      float mnew = fmaxf(rmax[e], mx);
      corr[e] = __expf(rmax[e] - mnew);
      rmax[e] = mnew;
      p[0][e] = __expf(sc0 - mnew);
      p[1][e] = __expf(sc1 - mnew);
      p[2][e] = __expf(sc2 - mnew);
      p[3][e] = __expf(sc3 - mnew);
      rsum[e] = rsum[e] * corr[e]
              + (p[0][e] + p[1][e] + p[2][e] + p[3][e]);
    }
#pragma unroll
    for (int ni = 0; ni < 4; ++ni)
#pragma unroll
      for (int e = 0; e < 8; ++e) oacc[ni][e] *= corr[e];

    // --- P: C-layout -> A-layout via wave-private LDS round-trip ---
#pragma unroll
    for (int g = 0; g < 4; ++g)
#pragma unroll
      for (int e = 0; e < 8; ++e)
        sP[wave][e + 8 * half][g * 16 + l16] = (bf16_t)p[g][e];
    v16bf pf[2];
#pragma unroll
    for (int c = 0; c < 2; ++c)
      pf[c] = lds_afrag(&sP[wave][l16][c * 32 + half * 8]);

    // --- O += P @ V (4 d-subtiles x 2 K-chunks) ---
#pragma unroll
    for (int ni = 0; ni < 4; ++ni)
#pragma unroll
      for (int c = 0; c < 2; ++c) {
        v16bf vf = lds_bfrag(&sVt[ni * 16 + l16][c * 32 + half * 8]);
        oacc[ni] = __builtin_amdgcn_wmma_f32_16x16x32_bf16(
            false, pf[c], false, vf, (short)0, oacc[ni], false, false);
      }
    __syncthreads();
  }

  // --- final: reduce lane-partial sums once, normalize, store bf16 ---
#pragma unroll
  for (int e = 0; e < 8; ++e) {
    float inv = 1.0f / rowsum16(rsum[e]);
    size_t tok = (size_t)b * SEQ + q0 + wave * 16 + e + 8 * half;
    bf16_t* op = out + tok * DIMC + h * HDIM;
#pragma unroll
    for (int ni = 0; ni < 4; ++ni)
      op[ni * 16 + l16] = (bf16_t)(oacc[ni][e] * inv);
  }
}

// ---------------------------------------------------------------------------
extern "C" void kernel_launch(void* const* d_in, const int* in_sizes, int n_in,
                              void* d_out, int out_size, void* d_ws, size_t ws_size,
                              hipStream_t stream)
{
  const float* x      = (const float*)d_in[0];
  const float* W_qkv  = (const float*)d_in[1];
  const float* b_qkv  = (const float*)d_in[2];
  const float* W_proj = (const float*)d_in[3];
  const float* b_proj = (const float*)d_in[4];
  float* out = (float*)d_out;

  char* ws = (char*)d_ws;
  bf16_t* xb    = (bf16_t*)ws;  ws += (size_t)NTOK * DIMC * sizeof(bf16_t);
  bf16_t* Wqkvb = (bf16_t*)ws;  ws += (size_t)DIMC * 3 * DIMC * sizeof(bf16_t);
  bf16_t* Wprjb = (bf16_t*)ws;  ws += (size_t)DIMC * DIMC * sizeof(bf16_t);
  bf16_t* qkvb  = (bf16_t*)ws;  ws += (size_t)NTOK * 3 * DIMC * sizeof(bf16_t);
  bf16_t* attb  = (bf16_t*)ws;

  {
    int n4;
    n4 = (NTOK * DIMC) / 4;
    pack_bf16_kernel<<<(n4 + 255) / 256, 256, 0, stream>>>(x, xb, n4);
    n4 = (DIMC * 3 * DIMC) / 4;
    pack_bf16_kernel<<<(n4 + 255) / 256, 256, 0, stream>>>(W_qkv, Wqkvb, n4);
    n4 = (DIMC * DIMC) / 4;
    pack_bf16_kernel<<<(n4 + 255) / 256, 256, 0, stream>>>(W_proj, Wprjb, n4);
  }

  gemm_bias_kernel<bf16_t><<<dim3((3 * DIMC) / 128, NTOK / 64), 128, 0, stream>>>(
      xb, Wqkvb, b_qkv, qkvb, NTOK, 3 * DIMC, DIMC);

  rope_kernel<<<(NTOK * 2 * NHEADS * 32) / 256, 256, 0, stream>>>(qkvb);

  attn_kernel<<<dim3(SEQ / 64, NHEADS, BATCH), 128, 0, stream>>>(qkvb, attb);

  gemm_bias_kernel<float><<<dim3(DIMC / 128, NTOK / 64), 128, 0, stream>>>(
      attb, Wprjb, b_proj, out, NTOK, DIMC, DIMC);
}
